// enhanced_temporal_contrastive_loss_64372969832691
// MI455X (gfx1250) — matrix operand order repaired and checked by reading, not compiled, not run on hardware
//
#include <hip/hip_runtime.h>
#include <hip/hip_bf16.h>

typedef _Float16 v16h __attribute__((ext_vector_type(16)));
typedef _Float16 v8h  __attribute__((ext_vector_type(8)));
typedef _Float16 v2h  __attribute__((ext_vector_type(2)));
typedef float    v8f  __attribute__((ext_vector_type(8)));

#define B_   64
#define S_   512
#define D_   768
#define INV_TEMP 10.0f          // 1 / TEMPERATURE
#define LDSA_STRIDE 776         // 768 + 8 halves pad: 388 dwords % 64 banks = 4 -> conflict-free

// ---------------- Kernel 1: L2-normalize rows, emit f16 ----------------
__global__ __launch_bounds__(256) void k_normalize(const float* __restrict__ x,
                                                   _Float16* __restrict__ xn) {
  const int wave = threadIdx.x >> 5;
  const int lane = threadIdx.x & 31;
  const int row  = blockIdx.x * 8 + wave;          // B_*S_ rows total
  const size_t base = (size_t)row * D_;

  float2 v[12];
  float ss = 0.f;
#pragma unroll
  for (int k = 0; k < 12; ++k) {
    v[k] = *(const float2*)(x + base + 2 * lane + 64 * k);
    ss += v[k].x * v[k].x + v[k].y * v[k].y;
  }
#pragma unroll
  for (int off = 16; off >= 1; off >>= 1) ss += __shfl_xor(ss, off, 32);
  const float inv = 1.0f / fmaxf(sqrtf(ss), 1e-12f);
#pragma unroll
  for (int k = 0; k < 12; ++k) {
    v2h h;
    h.x = (_Float16)(v[k].x * inv);
    h.y = (_Float16)(v[k].y * inv);
    *(v2h*)(xn + base + 2 * lane + 64 * k) = h;
  }
}

// ---------------- Kernel 2: fused Gram-matrix + contrastive loss ----------------
// grid = B_ * (S_/64) blocks; block = 256 threads (8 waves).
// Block owns (batch b, 64-row i-block) staged in LDS. Wave w owns j-tiles
// w*4 .. w*4+3, processed as TWO PAIRS: per k-chunk one A fragment (ds) feeds
// 2 WMMAs (two j-tiles), one B fragment (global) feeds 4 WMMAs (four i-tiles).
// Mix per k-chunk: 4 global_b128 + 8 ds_b128 + 8 wmma.
__global__ __launch_bounds__(256) void k_loss(const _Float16* __restrict__ xn,
                                              float* __restrict__ partials) {
  __shared__ _Float16 As[64 * LDSA_STRIDE];        // ~97 KB
  __shared__ float negPart[8 * 64];
  __shared__ float posLo[64], posHi[64], lossLds[64];

  const int tid  = threadIdx.x;
  const int wave = tid >> 5;
  const int lane = tid & 31;
  const int b    = blockIdx.x >> 3;        // 8 i-blocks per batch
  const int i0b  = (blockIdx.x & 7) << 6;  // 64-row i-block base

  // Stage A block (64 rows x 768 halves) into LDS; 4 threads per row, 384 B each.
  {
    const int rr = tid >> 2;
    const int c0 = (tid & 3) * 192;
    const _Float16* src = xn + ((size_t)(b * S_ + i0b + rr)) * D_ + c0;
    _Float16* dst = As + rr * LDSA_STRIDE + c0;
#pragma unroll
    for (int u = 0; u < 24; ++u)
      *(v8h*)(dst + u * 8) = *(const v8h*)(src + u * 8);
  }
  if (tid < 64) { posLo[tid] = 0.f; posHi[tid] = 0.f; }
  __syncthreads();

  const int hi   = lane >> 4;              // half-wave id (0/1)
  const int lo16 = lane & 15;              // N position / A row-in-tile

  float negAcc[4][8];
#pragma unroll
  for (int it = 0; it < 4; ++it)
#pragma unroll
    for (int r = 0; r < 8; ++r) negAcc[it][r] = 0.f;

  // A fragment (16x32 f16, ISA 7.12.2): lane holds row lo16 of its i-tile,
  // halves [kb + 8*hi .. +7] and [kb + 16 + 8*hi .. +7] -> two ds_load_b128.
  const _Float16* aRow[4];
#pragma unroll
  for (int it = 0; it < 4; ++it)
    aRow[it] = As + (it * 16 + lo16) * LDSA_STRIDE + 8 * hi;

  for (int jj = 0; jj < 2; ++jj) {         // two j-tile pairs per wave
    const int j0 = (wave * 4 + jj * 2) << 4;   // first tile of the pair
    // B fragment (32x16 f16): lane holds col j0+lo16, halves
    // [kb + 8*hi .. +7] and [kb + 16 + 8*hi .. +7] -> two global_load_b128 (L2-resident panel).
    const _Float16* bptr0 = xn + ((size_t)(b * S_ + j0 + lo16)) * D_ + 8 * hi;
    const _Float16* bptr1 = bptr0 + (size_t)16 * D_;   // j-tile j0+16

    v8f acc[2][4];
#pragma unroll
    for (int q = 0; q < 2; ++q)
#pragma unroll
      for (int it = 0; it < 4; ++it) acc[q][it] = (v8f){};

#pragma unroll 2
    for (int kk = 0; kk < 24; ++kk) {      // K = 768 = 24 * 32
      const int kb = kk * 32;
      v8h b00 = *(const v8h*)(bptr0 + kb);
      v8h b01 = *(const v8h*)(bptr0 + kb + 16);
      v8h b10 = *(const v8h*)(bptr1 + kb);
      v8h b11 = *(const v8h*)(bptr1 + kb + 16);
      v16h bf0 = __builtin_shufflevector(b00, b01, 0,1,2,3,4,5,6,7,8,9,10,11,12,13,14,15);
      v16h bf1 = __builtin_shufflevector(b10, b11, 0,1,2,3,4,5,6,7,8,9,10,11,12,13,14,15);
#pragma unroll
      for (int it = 0; it < 4; ++it) {
        v8h a0 = *(const v8h*)(aRow[it] + kb);
        v8h a1 = *(const v8h*)(aRow[it] + kb + 16);
        v16h af = __builtin_shufflevector(a0, a1, 0,1,2,3,4,5,6,7,8,9,10,11,12,13,14,15);
        acc[0][it] = __builtin_amdgcn_wmma_f32_16x16x32_f16(false, af, false, bf0,
                                                            (short)0, acc[0][it], false, false);
        acc[1][it] = __builtin_amdgcn_wmma_f32_16x16x32_f16(false, af, false, bf1,
                                                            (short)0, acc[1][it], false, false);
      }
    }

    // Epilogue: C layout -> (t = ti0 + r + 8*hi, j = j0q + lo16).
#pragma unroll
    for (int q = 0; q < 2; ++q) {
      const int j0q = j0 + q * 16;
      const int j   = j0q + lo16;
#pragma unroll
      for (int it = 0; it < 4; ++it) {
        const int ti0 = i0b + it * 16;
        // Uniform (scalar) tile classification: if every |j - t| >= 10 in this
        // 16x16 tile, take the branchless all-negatives path.
        if (j0q >= ti0 + 25 || j0q + 25 <= ti0) {
#pragma unroll
          for (int r = 0; r < 8; ++r)
            negAcc[it][r] += __expf(acc[q][it][r] * INV_TEMP);
        } else {
#pragma unroll
          for (int r = 0; r < 8; ++r) {
            const int tl = it * 16 + r + hi * 8;
            const int t  = i0b + tl;
            const float s = acc[q][it][r] * INV_TEMP;
            const int d  = j - t;
            const int ad = d < 0 ? -d : d;
            if (ad >= 10) negAcc[it][r] += __expf(s);   // NEGATIVE_WINDOW
            if (d == -1) posLo[tl] = s;                 // unique writer per t
            if (d ==  1) posHi[tl] = s;                 // unique writer per t
          }
        }
      }
    }
  }

  // Butterfly-reduce negAcc over the 16 lanes of each half-wave (j-direction).
#pragma unroll
  for (int it = 0; it < 4; ++it)
#pragma unroll
    for (int r = 0; r < 8; ++r) {
#pragma unroll
      for (int off = 8; off >= 1; off >>= 1)
        negAcc[it][r] += __shfl_xor(negAcc[it][r], off, 32);
    }
  if (lo16 == 0) {
#pragma unroll
    for (int it = 0; it < 4; ++it)
#pragma unroll
      for (int r = 0; r < 8; ++r)
        negPart[wave * 64 + it * 16 + hi * 8 + r] = negAcc[it][r];
  }
  __syncthreads();

  // Per-row loss: fixed-order sum over the 8 wave partials -> deterministic.
  if (tid < 64) {
    float ns = 0.f;
#pragma unroll
    for (int w2 = 0; w2 < 8; ++w2) ns += negPart[w2 * 64 + tid];
    const int t = i0b + tid;
    float loss = 0.f;
    if (t > 0)      { float s = posLo[tid]; loss += __logf(__expf(s) + ns) - s; }
    if (t < S_ - 1) { float s = posHi[tid]; loss += __logf(__expf(s) + ns) - s; }
    lossLds[tid] = loss;
  }
  __syncthreads();
  if (tid == 0) {
    float sum = 0.f;
#pragma unroll
    for (int u = 0; u < 64; ++u) sum += lossLds[u];
    partials[blockIdx.x] = sum;
  }
}

// ---------------- Kernel 3: deterministic final reduction ----------------
__global__ __launch_bounds__(256) void k_reduce(const float* __restrict__ partials,
                                                float* __restrict__ out) {
  __shared__ float red[256];
  float s = 0.f;
  for (int i = threadIdx.x; i < B_ * (S_ / 64); i += 256) s += partials[i];
  red[threadIdx.x] = s;
  __syncthreads();
  for (int off = 128; off >= 1; off >>= 1) {
    if (threadIdx.x < off) red[threadIdx.x] += red[threadIdx.x + off];
    __syncthreads();
  }
  if (threadIdx.x == 0) {
    // valid_pairs: every t has >=1 positive and >=1 negative at S=512;
    // pair count = 2*S-2 per batch.
    const float validPairs = (float)(B_ * (2 * S_ - 2));   // 64 * 1022
    out[0] = red[0] / validPairs;
  }
}

extern "C" void kernel_launch(void* const* d_in, const int* in_sizes, int n_in,
                              void* d_out, int out_size, void* d_ws, size_t ws_size,
                              hipStream_t stream) {
  const float* x = (const float*)d_in[0];
  _Float16* xn = (_Float16*)d_ws;                                   // 48 MiB f16 copy
  float* partials = (float*)((char*)d_ws +
                             (size_t)B_ * S_ * D_ * sizeof(_Float16));
  float* out = (float*)d_out;

  k_normalize<<<dim3((B_ * S_) / 8), dim3(256), 0, stream>>>(x, xn);
  k_loss<<<dim3(B_ * (S_ / 64)), dim3(256), 0, stream>>>(xn, partials);
  k_reduce<<<dim3(1), dim3(256), 0, stream>>>(partials, out);
}
